// InvariantPointAttentionB_33474975105103
// MI455X (gfx1250) — matrix operand order repaired and checked by reading, not compiled, not run on hardware
//
#include <hip/hip_runtime.h>
#include <hip/hip_bf16.h>
#include <math.h>

typedef __attribute__((ext_vector_type(16))) _Float16 v16h;
typedef __attribute__((ext_vector_type(8)))  float    v8f;
typedef __attribute__((ext_vector_type(4)))  float    f32x4;

#define NN   768
#define C1_  384
#define C2_  128
#define HH   12
#define SQK_ 16
#define SV_  16
#define PQK_ 4
#define PV_  8
#define NC_  384

__device__ __constant__ const float SCALAR_W = 0.14433756729740643f;  // sqrt(1/48)
__device__ __constant__ const float POINT_W  = 0.13608276348795434f;  // sqrt(1/54)
__device__ __constant__ const float W2D_     = 0.5773502691896258f;   // sqrt(1/3)
__device__ __constant__ const float EPS_     = 1e-8f;

__device__ __forceinline__ float softplusf(float x) {
    return (x > 20.0f) ? x : log1pf(__expf(x));
}

// ---------- WMMA fragment loaders (f32 -> f16 on the fly) ----------
// A 16x32 f16: lane L holds row (L&15); half = L>>4 selects K-groups
//   elements 0..15 -> K = 16*half + 0..15 (contiguous)
__device__ __forceinline__ v16h load_a_row(const float* __restrict__ rowp, int k0, int half) {
    const f32x4* p = (const f32x4*)(rowp + k0 + 16 * half);
    const f32x4* q = (const f32x4*)(rowp + k0 + 16 * half + 8);
    f32x4 x0 = p[0], x1 = p[1], y0 = q[0], y1 = q[1];
    v16h v;
#pragma unroll
    for (int i = 0; i < 4; ++i) {
        v[i]      = (_Float16)x0[i];
        v[4 + i]  = (_Float16)x1[i];
        v[8 + i]  = (_Float16)y0[i];
        v[12 + i] = (_Float16)y1[i];
    }
    return v;
}
__device__ __forceinline__ v16h zero16() { v16h v = {}; return v; }

__device__ __forceinline__ v16h load_a_row_g(const float* __restrict__ rowp, int k0, int half, bool valid) {
    if (!valid) return zero16();
    return load_a_row(rowp, k0, half);
}

// B 32x16 f16: lane L holds col (L&15); elements 0..15 -> K = 16*half + 0..15 (contiguous)
__device__ __forceinline__ v16h load_b_row(const float* __restrict__ rowp, int k0, int half) {
    const f32x4* p = (const f32x4*)(rowp + k0 + 16 * half);
    f32x4 x0 = p[0], x1 = p[1], x2 = p[2], x3 = p[3];
    v16h v;
#pragma unroll
    for (int i = 0; i < 4; ++i) {
        v[i]      = (_Float16)x0[i];
        v[4 + i]  = (_Float16)x1[i];
        v[8 + i]  = (_Float16)x2[i];
        v[12 + i] = (_Float16)x3[i];
    }
    return v;
}
__device__ __forceinline__ v16h load_b_row_g(const float* __restrict__ rowp, int k0, int half, bool valid) {
    if (!valid) return zero16();
    return load_b_row(rowp, k0, half);
}

#define WMMA_F16(a, b, c) \
    __builtin_amdgcn_wmma_f32_16x16x32_f16(false, (a), false, (b), (short)0, (c), false, false)

// ---------- generic Y[M][Nout] = X[M][K] @ W[Nout][K]^T + bias ----------
__global__ __launch_bounds__(32) void wmma_gemm_xwT(const float* __restrict__ X,
                                                    const float* __restrict__ W,
                                                    const float* __restrict__ bias,
                                                    float* __restrict__ Y,
                                                    int Nout, int K) {
    int tiles_n = Nout >> 4;
    int ti = blockIdx.x / tiles_n, tj = blockIdx.x % tiles_n;
    int lane = threadIdx.x & 31, half = lane >> 4, l15 = lane & 15;
    const float* Arow = X + (size_t)(ti * 16 + l15) * K;
    const float* Brow = W + (size_t)(tj * 16 + l15) * K;
    v8f acc = {};
    for (int k0 = 0; k0 < K; k0 += 32) {
        v16h a = load_a_row(Arow, k0, half);
        v16h b = load_b_row(Brow, k0, half);
        acc = WMMA_F16(a, b, acc);
    }
    float bv = bias[tj * 16 + l15];
#pragma unroll
    for (int r = 0; r < 8; ++r) {
        int m = ti * 16 + r + 8 * half;
        Y[(size_t)m * Nout + tj * 16 + l15] = acc[r] + bv;
    }
}

// ---------- prep: rotate points, pack fused QK vectors, split k/v (transposed V) ----------
__global__ void prep_kernel(const float* __restrict__ qs, const float* __restrict__ kvs,
                            const float* __restrict__ qp, const float* __restrict__ kvp,
                            const float* __restrict__ rot, const float* __restrict__ trans,
                            const float* __restrict__ tpw,
                            float* __restrict__ qA, float* __restrict__ kB,
                            float* __restrict__ qn, float* __restrict__ kn,
                            float* __restrict__ vS_T, float* __restrict__ vP_T) {
    int idx = blockIdx.x * blockDim.x + threadIdx.x;
    if (idx >= NN * HH) return;
    int n = idx / HH, h = idx % HH;
    float R[9];
#pragma unroll
    for (int c = 0; c < 9; ++c) R[c] = rot[n * 9 + c];
    float t0 = trans[n * 3 + 0], t1 = trans[n * 3 + 1], t2 = trans[n * 3 + 2];
    float pwv = softplusf(tpw[h]) * POINT_W;

    float* qAp = qA + ((size_t)h * NN + n) * 32;
#pragma unroll
    for (int d = 0; d < 16; ++d) qAp[d] = SCALAR_W * qs[(size_t)n * 192 + h * 16 + d];
    float qacc = 0.f;
#pragma unroll
    for (int p = 0; p < PQK_; ++p) {
        float a0 = qp[(size_t)n * 144 + 0 * 48 + h * 4 + p];
        float a1 = qp[(size_t)n * 144 + 1 * 48 + h * 4 + p];
        float a2 = qp[(size_t)n * 144 + 2 * 48 + h * 4 + p];
        float g0 = R[0] * a0 + R[1] * a1 + R[2] * a2 + t0;
        float g1 = R[3] * a0 + R[4] * a1 + R[5] * a2 + t1;
        float g2 = R[6] * a0 + R[7] * a1 + R[8] * a2 + t2;
        qAp[16 + p * 3 + 0] = pwv * g0;
        qAp[16 + p * 3 + 1] = pwv * g1;
        qAp[16 + p * 3 + 2] = pwv * g2;
        qacc += g0 * g0 + g1 * g1 + g2 * g2;
    }
    qAp[28] = qAp[29] = qAp[30] = qAp[31] = 0.f;
    qn[h * NN + n] = qacc;

    float* kBp = kB + ((size_t)h * NN + n) * 32;
#pragma unroll
    for (int d = 0; d < 16; ++d) {
        kBp[d] = kvs[(size_t)n * 384 + h * 32 + d];
        // transposed: vS_T[h][d][n]
        vS_T[((size_t)h * 16 + d) * NN + n] = kvs[(size_t)n * 384 + h * 32 + 16 + d];
    }
    float kacc = 0.f;
#pragma unroll
    for (int p = 0; p < 12; ++p) {
        float a0 = kvp[(size_t)n * 432 + 0 * 144 + h * 12 + p];
        float a1 = kvp[(size_t)n * 432 + 1 * 144 + h * 12 + p];
        float a2 = kvp[(size_t)n * 432 + 2 * 144 + h * 12 + p];
        float g0 = R[0] * a0 + R[1] * a1 + R[2] * a2 + t0;
        float g1 = R[3] * a0 + R[4] * a1 + R[5] * a2 + t1;
        float g2 = R[6] * a0 + R[7] * a1 + R[8] * a2 + t2;
        if (p < 4) {
            kBp[16 + p * 3 + 0] = g0;
            kBp[16 + p * 3 + 1] = g1;
            kBp[16 + p * 3 + 2] = g2;
            kacc += g0 * g0 + g1 * g1 + g2 * g2;
        } else {
            // transposed: vP_T[h][(p-4)*3+c][n]
            vP_T[((size_t)h * 24 + (p - 4) * 3 + 0) * NN + n] = g0;
            vP_T[((size_t)h * 24 + (p - 4) * 3 + 1) * NN + n] = g1;
            vP_T[((size_t)h * 24 + (p - 4) * 3 + 2) * NN + n] = g2;
        }
    }
    kBp[28] = kBp[29] = kBp[30] = kBp[31] = 0.f;
    kn[h * NN + n] = kacc;
}

// ---------- a2d: attn[h][i][j] = W2D*(inputs_2d[i][j][:] . w_a2[h][:] + b_a2[h]) ----------
__global__ __launch_bounds__(32) void a2d_kernel(const float* __restrict__ in2d,
                                                 const float* __restrict__ w_a2,
                                                 const float* __restrict__ b_a2,
                                                 float* __restrict__ attn) {
    int i = blockIdx.x / 48, jt = blockIdx.x % 48;
    int lane = threadIdx.x & 31, half = lane >> 4, l15 = lane & 15;
    const float* Arow = in2d + ((size_t)i * NN + jt * 16 + l15) * C2_;   // row j, 128 c
    bool bvalid = (l15 < HH);
    const float* Brow = w_a2 + (size_t)l15 * C2_;                        // row h
    v8f acc = {};
#pragma unroll
    for (int k0 = 0; k0 < C2_; k0 += 32) {
        if (k0 + 32 < C2_) __builtin_prefetch(Arow + k0 + 32, 0, 0);
        v16h a = load_a_row(Arow, k0, half);
        v16h b = load_b_row_g(Brow, k0, half, bvalid);
        acc = WMMA_F16(a, b, acc);
    }
    if (l15 < HH) {
        int h = l15;
        float bb = b_a2[h];
#pragma unroll
        for (int r = 0; r < 8; ++r) {
            int j = jt * 16 + r + 8 * half;
            attn[((size_t)h * NN + i) * NN + j] = W2D_ * (acc[r] + bb);
        }
    }
}

// ---------- logits: fused scalar-QK + point-QK (K=28 padded to 32), add biases ----------
__global__ __launch_bounds__(32) void logits_kernel(const float* __restrict__ qA,
                                                    const float* __restrict__ kB,
                                                    const float* __restrict__ qn,
                                                    const float* __restrict__ kn,
                                                    const float* __restrict__ tpw,
                                                    const float* __restrict__ mask,
                                                    float* __restrict__ attn) {
    int h = blockIdx.x / (48 * 48);
    int rem = blockIdx.x % (48 * 48);
    int it = rem / 48, jt = rem % 48;
    int lane = threadIdx.x & 31, half = lane >> 4, l15 = lane & 15;
    float pwv = softplusf(tpw[h]) * POINT_W;
    const float* Arow = qA + ((size_t)h * NN + it * 16 + l15) * 32;
    const float* Brow = kB + ((size_t)h * NN + jt * 16 + l15) * 32;
    v16h a = load_a_row(Arow, 0, half);
    v16h b = load_b_row(Brow, 0, half);
    v8f acc = {};
    acc = WMMA_F16(a, b, acc);
    int j = jt * 16 + l15;
    float knj = kn[h * NN + j];
    float mj = mask[j];
#pragma unroll
    for (int r = 0; r < 8; ++r) {
        int i = it * 16 + r + 8 * half;
        size_t idx = ((size_t)h * NN + i) * NN + j;
        float v = acc[r] - 0.5f * pwv * (qn[h * NN + i] + knj);
        v += attn[idx];                                // W2D * a2d (pre-stored)
        v -= 100000.0f * (1.0f - mask[i] * mj);
        attn[idx] = v;
    }
}

// ---------- softmax over j (768) per (h,i) row, one wave32 per row ----------
__global__ __launch_bounds__(32) void softmax_kernel(float* __restrict__ attn) {
    size_t row = blockIdx.x;
    float* p = attn + row * NN;
    int lane = threadIdx.x;
    float mx = -1e30f;
    for (int j = lane; j < NN; j += 32) mx = fmaxf(mx, p[j]);
#pragma unroll
    for (int o = 16; o > 0; o >>= 1) mx = fmaxf(mx, __shfl_xor(mx, o, 32));
    float s = 0.f;
    for (int j = lane; j < NN; j += 32) {
        float e = __expf(p[j] - mx);
        p[j] = e;
        s += e;
    }
#pragma unroll
    for (int o = 16; o > 0; o >>= 1) s += __shfl_xor(s, o, 32);
    float inv = 1.0f / s;
    for (int j = lane; j < NN; j += 32) p[j] *= inv;
}

// ---------- attn @ v_scalar (16 cols) and attn @ v_point (24 cols, 2 tiles) ----------
// V matrices are stored transposed ([h][col][j]) so B fragments are contiguous loads.
__global__ __launch_bounds__(32) void attn_v_kernel(const float* __restrict__ attn,
                                                    const float* __restrict__ vS_T,
                                                    const float* __restrict__ vP_T,
                                                    float* __restrict__ fact,
                                                    float* __restrict__ rpg) {
    int h = blockIdx.x / 48, it = blockIdx.x % 48;
    int lane = threadIdx.x & 31, half = lane >> 4, l15 = lane & 15;
    const float* Arow = attn + ((size_t)h * NN + it * 16 + l15) * NN;
    const float* vSb = vS_T + (size_t)h * 16 * NN;
    const float* vPb = vP_T + (size_t)h * 24 * NN;
    const float* b1row = vSb + (size_t)l15 * NN;
    const float* b2row = vPb + (size_t)l15 * NN;
    const float* b3row = vPb + (size_t)(16 + l15) * NN;   // only deref'd when l15 < 8
    v8f a1 = {}, a2 = {}, a3 = {};
    for (int k0 = 0; k0 < NN; k0 += 32) {
        v16h a = load_a_row(Arow, k0, half);
        v16h b1 = load_b_row(b1row, k0, half);
        v16h b2 = load_b_row(b2row, k0, half);
        v16h b3 = load_b_row_g(b3row, k0, half, l15 < 8);
        a1 = WMMA_F16(a, b1, a1);
        a2 = WMMA_F16(a, b2, a2);
        a3 = WMMA_F16(a, b3, a3);
    }
#pragma unroll
    for (int r = 0; r < 8; ++r) {
        int i = it * 16 + r + 8 * half;
        fact[(size_t)i * 2112 + h * 16 + l15] = a1[r];                 // result_scalar
        rpg[((size_t)h * NN + i) * 24 + l15] = a2[r];
        if (l15 < 8) rpg[((size_t)h * NN + i) * 24 + 16 + l15] = a3[r];
    }
}

// ---------- invert frames: local = R^T (rpg - t), dist ----------
__global__ void finalize_points(const float* __restrict__ rpg,
                                const float* __restrict__ rot,
                                const float* __restrict__ trans,
                                float* __restrict__ fact) {
    int idx = blockIdx.x * blockDim.x + threadIdx.x;
    if (idx >= NN * HH * PV_) return;
    int n = idx / (HH * PV_);
    int m = idx % (HH * PV_);
    int h = m / PV_, v = m % PV_;
    float g0 = rpg[((size_t)h * NN + n) * 24 + v * 3 + 0] - trans[n * 3 + 0];
    float g1 = rpg[((size_t)h * NN + n) * 24 + v * 3 + 1] - trans[n * 3 + 1];
    float g2 = rpg[((size_t)h * NN + n) * 24 + v * 3 + 2] - trans[n * 3 + 2];
    float R[9];
#pragma unroll
    for (int c = 0; c < 9; ++c) R[c] = rot[n * 9 + c];
    float l0 = R[0] * g0 + R[3] * g1 + R[6] * g2;
    float l1 = R[1] * g0 + R[4] * g1 + R[7] * g2;
    float l2 = R[2] * g0 + R[5] * g1 + R[8] * g2;
    float* fa = fact + (size_t)n * 2112;
    fa[192 + m] = l0;
    fa[288 + m] = l1;
    fa[384 + m] = l2;
    fa[480 + m] = sqrtf(EPS_ + l0 * l0 + l1 * l1 + l2 * l2);
}

// ---------- r2d: per i, out[h][c] = attn[h][i][:] @ inputs_2d[i][:][c] ----------
// 256-thread block per i. The 32x128 f32 inputs_2d tile is staged into LDS with
// gfx1250 async loads (global_load_async_to_lds_b128, ASYNCcnt), double-buffered.
// Wave w (0..7) owns c-tile w and feeds WMMA from LDS.
__global__ __launch_bounds__(256) void attn_2d_kernel(const float* __restrict__ attn,
                                                      const float* __restrict__ in2d,
                                                      float* __restrict__ fact) {
    __shared__ float tile[2][32 * C2_];   // 2 x 16 KB
    int i = blockIdx.x;
    int t = threadIdx.x;
    int ct = t >> 5;                       // wave id -> c-tile
    int lane = t & 31, half = lane >> 4, l15 = lane & 15;
    bool avalid = (l15 < HH);
    const float* Arow = attn + ((size_t)l15 * NN + i) * NN;            // row h = l15
    const float* Bbase = in2d + (size_t)i * NN * C2_;

    // Each thread copies 4 x b128 (64 B) per K-step: 256 thr * 64 B = 16 KB tile.
    auto issue_tile = [&](int k0, int buf) {
#pragma unroll
        for (int s = 0; s < 4; ++s) {
            int q = t * 4 + s;                 // 0..1023 float4 chunks
            int j = q >> 5;                    // 0..31 (row within k-slab)
            int cq = (q & 31) * 4;             // 0..124 (col)
            const float* g = Bbase + (size_t)(k0 + j) * C2_ + cq;
            unsigned lds = (unsigned)(size_t)&tile[buf][(q << 2)];
            asm volatile("global_load_async_to_lds_b128 %0, %1, off"
                         :: "v"(lds), "v"(g) : "memory");
        }
    };

    issue_tile(0, 0);
    v8f acc = {};
    for (int k0 = 0; k0 < NN; k0 += 32) {
        int buf = (k0 >> 5) & 1;
        asm volatile("s_wait_asynccnt 0x0" ::: "memory");
        __syncthreads();                       // whole tile visible to all waves
        if (k0 + 32 < NN) issue_tile(k0 + 32, buf ^ 1);
        v16h a = load_a_row_g(Arow, k0, half, avalid);
        v16h b;
        const float* tb = &tile[buf][0];
#pragma unroll
        for (int e = 0; e < 16; ++e)
            b[e] = (_Float16)tb[(16 * half + e) * C2_ + ct * 16 + l15];
        acc = WMMA_F16(a, b, acc);
    }
#pragma unroll
    for (int r = 0; r < 8; ++r) {
        int h = r + 8 * half;
        if (h < HH)
            fact[(size_t)i * 2112 + 576 + h * 128 + ct * 16 + l15] = acc[r];
    }
}

extern "C" void kernel_launch(void* const* d_in, const int* in_sizes, int n_in,
                              void* d_out, int out_size, void* d_ws, size_t ws_size,
                              hipStream_t stream) {
    const float* in1d  = (const float*)d_in[0];
    const float* in2d  = (const float*)d_in[1];
    const float* mask  = (const float*)d_in[2];
    const float* rot   = (const float*)d_in[3];
    const float* trans = (const float*)d_in[4];
    const float* w_qs  = (const float*)d_in[5];
    const float* b_qs  = (const float*)d_in[6];
    const float* w_kvs = (const float*)d_in[7];
    const float* b_kvs = (const float*)d_in[8];
    const float* w_qp  = (const float*)d_in[9];
    const float* b_qp  = (const float*)d_in[10];
    const float* w_kvp = (const float*)d_in[11];
    const float* b_kvp = (const float*)d_in[12];
    const float* tpw   = (const float*)d_in[13];
    const float* w_a2  = (const float*)d_in[14];
    const float* b_a2  = (const float*)d_in[15];
    const float* w_o   = (const float*)d_in[16];
    const float* b_o   = (const float*)d_in[17];
    float* out = (float*)d_out;

    float* ws = (float*)d_ws;
    float* proj_qs  = ws;                               // 768*192
    float* proj_kvs = proj_qs  + (size_t)768 * 192;     // 768*384
    float* proj_qp  = proj_kvs + (size_t)768 * 384;     // 768*144
    float* proj_kvp = proj_qp  + (size_t)768 * 144;     // 768*432
    float* qA   = proj_kvp + (size_t)768 * 432;         // 12*768*32
    float* kB   = qA  + (size_t)12 * 768 * 32;          // 12*768*32
    float* qn   = kB  + (size_t)12 * 768 * 32;          // 12*768
    float* kn   = qn  + (size_t)12 * 768;               // 12*768
    float* vS_T = kn  + (size_t)12 * 768;               // 12*16*768
    float* vP_T = vS_T + (size_t)12 * 16 * 768;         // 12*24*768
    float* attn = vP_T + (size_t)12 * 24 * 768;         // 12*768*768
    float* rpg  = attn + (size_t)12 * 768 * 768;        // 12*768*24
    float* fact = rpg + (size_t)12 * 768 * 24;          // 768*2112

    dim3 b32(32);
    // 1) input projections (WMMA GEMMs)
    wmma_gemm_xwT<<<48 * 12, b32, 0, stream>>>(in1d, w_qs,  b_qs,  proj_qs,  192, 384);
    wmma_gemm_xwT<<<48 * 24, b32, 0, stream>>>(in1d, w_kvs, b_kvs, proj_kvs, 384, 384);
    wmma_gemm_xwT<<<48 * 9,  b32, 0, stream>>>(in1d, w_qp,  b_qp,  proj_qp,  144, 384);
    wmma_gemm_xwT<<<48 * 27, b32, 0, stream>>>(in1d, w_kvp, b_kvp, proj_kvp, 432, 384);
    // 2) rotate points, pack fused QK vectors, transposed V
    prep_kernel<<<(768 * 12 + 255) / 256, 256, 0, stream>>>(proj_qs, proj_kvs, proj_qp,
                                                            proj_kvp, rot, trans, tpw,
                                                            qA, kB, qn, kn, vS_T, vP_T);
    // 3) 2D attention bias (reads inputs_2d once)
    a2d_kernel<<<768 * 48, b32, 0, stream>>>(in2d, w_a2, b_a2, attn);
    // 4) logits via single fused WMMA + softmax
    logits_kernel<<<12 * 48 * 48, b32, 0, stream>>>(qA, kB, qn, kn, tpw, mask, attn);
    softmax_kernel<<<12 * 768, b32, 0, stream>>>(attn);
    // 5) attention-weighted outputs
    attn_v_kernel<<<12 * 48, b32, 0, stream>>>(attn, vS_T, vP_T, fact, rpg);
    finalize_points<<<(768 * 96 + 255) / 256, 256, 0, stream>>>(rpg, rot, trans, fact);
    attn_2d_kernel<<<768, 256, 0, stream>>>(attn, in2d, fact);
    // 6) output projection
    wmma_gemm_xwT<<<48 * 24, b32, 0, stream>>>(fact, w_o, b_o, out, 384, 2112);
}
